// Quantizer_25778393711180
// MI455X (gfx1250) — matrix operand order repaired and checked by reading, not compiled, not run on hardware
//
#include <hip/hip_runtime.h>

typedef __attribute__((ext_vector_type(16))) _Float16 v16h;
typedef __attribute__((ext_vector_type(8)))  _Float16 v8h;
typedef __attribute__((ext_vector_type(4)))  _Float16 v4h;
typedef __attribute__((ext_vector_type(8)))  float    v8f;
typedef __attribute__((ext_vector_type(4)))  float    v4f;

#define K_ENTRIES   8192
#define D_DIM       256
#define N_ROWS      32768          // 32 * 32 * 32
#define HW          1024           // 32*32 spatial per batch
#define ZB_STRIDE   262144         // 256*1024 floats per batch
#define ROWS_PER_WG 128
#define RSH         264            // LDS row stride in halfs (256 + 8 pad -> conflict-free b128)
#define CB_SCALE    4096.0f
#define LO_SCALE    2048.0f

// LDS plan (halfs): zhi[128*264]=33792, zlo @33792, bhi @67584 (16*264), blo @71808
#define ZLO_OFF 33792
#define BHI_OFF 67584
#define BLO_OFF 71808
#define SMEM_BYTES (76032 * 2)     // 152064 B -> 2 WGs per WGP (<= 160 KB each)

// AMDGPU sched_group_barrier masks
#define SGB_MFMA_WMMA 0x008
#define SGB_DS_READ   0x200

static __device__ __forceinline__ v16h cat8(v8h a, v8h b) {
  return __builtin_shufflevector(a, b, 0,1,2,3,4,5,6,7,8,9,10,11,12,13,14,15);
}

// ---------------- kernel 0: scaled codebook norms -------------------------
__global__ __launch_bounds__(256) void vq_cnorm(const float* __restrict__ cb,
                                                float* __restrict__ cn) {
  __shared__ float red[256];
  const int e = blockIdx.x, tid = threadIdx.x;
  const float v = cb[(size_t)e * D_DIM + tid];
  red[tid] = v * v;
  __syncthreads();
  for (int s = 128; s > 0; s >>= 1) {
    if (tid < s) red[tid] += red[tid + s];
    __syncthreads();
  }
  if (tid == 0) cn[e] = red[0] * CB_SCALE;   // 4096 * ||c||^2
}

// ---------------- kernel 1: fused split-f16 WMMA GEMM + argmin ------------
__global__ __launch_bounds__(256) void vq_argmin(const float* __restrict__ z,
                                                 const float* __restrict__ cb,
                                                 const float* __restrict__ cn,
                                                 int* __restrict__ idx_out) {
  extern __shared__ __align__(16) char smem_raw[];
  _Float16* lds = (_Float16*)smem_raw;
  _Float16* zhi = lds;
  _Float16* zlo = lds + ZLO_OFF;
  _Float16* bhi = lds + BHI_OFF;
  _Float16* blo = lds + BLO_OFF;

  const int tid  = threadIdx.x;
  const int wave = tid >> 5;
  const int lane = tid & 31;
  const int half = lane >> 4;          // which 16-lane half of the wave32
  const int m    = lane & 15;

  const int n0  = blockIdx.x * ROWS_PER_WG;   // 128 | 1024 -> single batch b
  const int b   = n0 >> 10;
  const int hw0 = n0 & 1023;

  // ---- one-time: stage 128 z rows as f16 hi/lo planes (coalesced loads) ----
  {
    const int tx = tid & 127;          // row within WG
    const int ty = tid >> 7;           // 0..1
    const float* zb = z + (size_t)b * ZB_STRIDE + hw0 + tx;
    _Float16* zh = zhi + tx * RSH;
    _Float16* zl = zlo + tx * RSH;
    for (int dbase = ty * 4; dbase < D_DIM; dbase += 8) {
      v4h hv, lv;
#pragma unroll
      for (int q = 0; q < 4; ++q) {
        float v = zb[(size_t)(dbase + q) * HW];
        _Float16 h = (_Float16)v;
        hv[q] = h;
        lv[q] = (_Float16)((v - (float)h) * LO_SCALE);
      }
      *(v4h*)(zh + dbase) = hv;
      *(v4h*)(zl + dbase) = lv;
    }
  }

  const _Float16* arow_h = zhi + (wave * 16 + m) * RSH;
  const _Float16* arow_l = zlo + (wave * 16 + m) * RSH;
  const _Float16* brow_h = bhi + m * RSH;
  const _Float16* brow_l = blo + m * RSH;

  // B-staging role of this thread: entry se, dword-group sd4
  const int se  = tid >> 4;            // 0..15 codebook entry within tile
  const int sd4 = tid & 15;
  _Float16* sbh = bhi + se * RSH;
  _Float16* sbl = blo + se * RSH;

  float minv[8];
  int   mini[8];
#pragma unroll
  for (int r = 0; r < 8; ++r) { minv[r] = 3.4e38f; mini[r] = 0x7fffffff; }

  // prefetch codebook tile 0 into registers
  v4f pref[4];
#pragma unroll
  for (int jj = 0; jj < 4; ++jj)
    pref[jj] = *(const v4f*)(cb + (size_t)se * D_DIM + (sd4 + 16 * jj) * 4);

  for (int t = 0; t < K_ENTRIES / 16; ++t) {
    __syncthreads();                   // prior pass's compute done (t=0: z staging done)
    // ---- store prefetched tile t (scaled, split hi/lo) into LDS ----
#pragma unroll
    for (int jj = 0; jj < 4; ++jj) {
      const int d = (sd4 + 16 * jj) * 4;
      v4h hv, lv;
#pragma unroll
      for (int q = 0; q < 4; ++q) {
        float cv = pref[jj][q] * CB_SCALE;
        _Float16 h = (_Float16)cv;
        hv[q] = h;
        lv[q] = (_Float16)((cv - (float)h) * LO_SCALE);
      }
      *(v4h*)(sbh + d) = hv;
      *(v4h*)(sbl + d) = lv;
    }
    // ---- issue global prefetch of tile t+1 (overlaps with compute below) ----
    if (t + 1 < K_ENTRIES / 16) {
      const float* crow = cb + (size_t)((t + 1) * 16 + se) * D_DIM;
#pragma unroll
      for (int jj = 0; jj < 4; ++jj)
        pref[jj] = *(const v4f*)(crow + (sd4 + 16 * jj) * 4);
    }
    __syncthreads();                   // tile t visible to all waves

    // ---- WMMA k-loop: 8 steps x (8 ds_load_b128 + 3 wmma) ----
    v8f acc1 = {};   // zh*ch
    v8f acc2 = {};   // 2048*(zh*cl)
    v8f acc3 = {};   // 2048*(zl*ch)
#pragma unroll
    for (int ki = 0; ki < 8; ++ki) {
      const int k  = ki * 32;
      const int a0 = k + 8 * half;     // A: elems 0-7 -> K=k+8h, elems 8-15 -> K=k+16+8h
      const int b0 = k + 8 * half;     // B: elems 0-7 -> K=k+8h, elems 8-15 -> K=k+16+8h, column n = lane&15
      v16h Ah = cat8(*(const v8h*)(arow_h + a0), *(const v8h*)(arow_h + a0 + 16));
      v16h Al = cat8(*(const v8h*)(arow_l + a0), *(const v8h*)(arow_l + a0 + 16));
      v16h Bh = cat8(*(const v8h*)(brow_h + b0), *(const v8h*)(brow_h + b0 + 16));
      v16h Bl = cat8(*(const v8h*)(brow_l + b0), *(const v8h*)(brow_l + b0 + 16));
      acc1 = __builtin_amdgcn_wmma_f32_16x16x32_f16(false, Ah, false, Bh, (short)0, acc1, false, false);
      acc2 = __builtin_amdgcn_wmma_f32_16x16x32_f16(false, Ah, false, Bl, (short)0, acc2, false, false);
      acc3 = __builtin_amdgcn_wmma_f32_16x16x32_f16(false, Al, false, Bh, (short)0, acc3, false, false);
    }
#if __has_builtin(__builtin_amdgcn_sched_group_barrier)
    // Skewed schedule: loads of step k+1 issue before WMMAs of step k, so the
    // s_wait_dscnt ahead of each WMMA group leaves the next 8 loads in flight.
    __builtin_amdgcn_sched_group_barrier(SGB_DS_READ, 8, 0);     // loads k=0
#pragma unroll
    for (int i = 0; i < 7; ++i) {
      __builtin_amdgcn_sched_group_barrier(SGB_DS_READ, 8, 0);   // loads k=i+1
      __builtin_amdgcn_sched_group_barrier(SGB_MFMA_WMMA, 3, 0); // wmmas k=i
    }
    __builtin_amdgcn_sched_group_barrier(SGB_MFMA_WMMA, 3, 0);   // wmmas k=7
#endif

    const int   j   = t * 16 + m;
    const float cnv = cn[j];           // 4096*||c_j||^2
#pragma unroll
    for (int r = 0; r < 8; ++r) {
      float dot = acc1[r] + (acc2[r] + acc3[r]) * (1.0f / LO_SCALE);  // = 4096 * z.c_j
      float s   = cnv - 2.0f * dot;                                   // 4096*(dist - ||z||^2)
      if (s < minv[r] || (s == minv[r] && j < mini[r])) { minv[r] = s; mini[r] = j; }
    }
  }

  // ---- butterfly argmin across the 16 lanes of each half ----
#pragma unroll
  for (int r = 0; r < 8; ++r) {
    float v = minv[r];
    int   i = mini[r];
#pragma unroll
    for (int mask = 1; mask < 16; mask <<= 1) {
      float ov = __shfl_xor(v, mask, 32);
      int   oi = __shfl_xor(i, mask, 32);
      if (ov < v || (ov == v && oi < i)) { v = ov; i = oi; }
    }
    minv[r] = v; mini[r] = i;
  }
  if (m == 0) {
#pragma unroll
    for (int r = 0; r < 8; ++r) {
      const int row = n0 + wave * 16 + half * 8 + r;   // C/D layout: VGPR r, half h -> M = r + 8h
      idx_out[row] = mini[r];
    }
  }
}

// ---------------- kernel 2: gather + transpose + partial MSE --------------
__global__ __launch_bounds__(256) void vq_gather(const float* __restrict__ z,
                                                 const float* __restrict__ cb,
                                                 const int* __restrict__ idx,
                                                 float* __restrict__ out,
                                                 float* __restrict__ partial) {
  __shared__ float red[256];
  const int tid = threadIdx.x;
  const int tx  = tid & 63, ty = tid >> 6;
  const int n0  = blockIdx.x * 64;          // 64 | 1024 -> single batch b
  const int bb  = n0 >> 10;
  const int hw  = (n0 & 1023) + tx;
  const int j   = idx[n0 + tx];
  const float* q  = cb  + (size_t)j * D_DIM;
  const float* zp = z   + (size_t)bb * ZB_STRIDE + hw;
  float*       op = out + (size_t)bb * ZB_STRIDE + hw;
  float sum = 0.0f;
  for (int d = ty; d < D_DIM; d += 4) {
    float zv = zp[(size_t)d * HW];
    float qv = q[d];
    float df = zv - qv;
    sum += df * df;
    op[(size_t)d * HW] = zv + (qv - zv);    // straight-through value == quantized
  }
  red[tid] = sum;
  __syncthreads();
  for (int s = 128; s > 0; s >>= 1) {
    if (tid < s) red[tid] += red[tid + s];
    __syncthreads();
  }
  if (tid == 0) partial[blockIdx.x] = red[0];
}

// ---------------- kernel 3: finalize losses -------------------------------
__global__ __launch_bounds__(256) void vq_finalize(const float* __restrict__ partial,
                                                   float* __restrict__ losses) {
  __shared__ float red[256];
  const int tid = threadIdx.x;
  red[tid] = partial[tid] + partial[tid + 256];   // 512 partials
  __syncthreads();
  for (int s = 128; s > 0; s >>= 1) {
    if (tid < s) red[tid] += red[tid + s];
    __syncthreads();
  }
  if (tid == 0) {
    float mean = red[0] / (float)(N_ROWS * D_DIM);
    losses[0] = mean;           // codebook loss
    losses[1] = 0.2f * mean;    // commitment loss = BETA * same value
  }
}

extern "C" void kernel_launch(void* const* d_in, const int* in_sizes, int n_in,
                              void* d_out, int out_size, void* d_ws, size_t ws_size,
                              hipStream_t stream) {
  const float* z  = (const float*)d_in[0];
  const float* cb = (const float*)d_in[1];
  float* out = (float*)d_out;

  char*  ws      = (char*)d_ws;
  float* cn      = (float*)ws;                              // 8192 f32
  int*   idx     = (int*)(ws + 8192 * 4);                   // 32768 i32
  float* partial = (float*)(ws + 8192 * 4 + 32768 * 4);     // 512 f32

  (void)hipFuncSetAttribute((const void*)vq_argmin,
                            hipFuncAttributeMaxDynamicSharedMemorySize, SMEM_BYTES);

  vq_cnorm   <<<K_ENTRIES, 256, 0, stream>>>(cb, cn);
  vq_argmin  <<<N_ROWS / ROWS_PER_WG, 256, SMEM_BYTES, stream>>>(z, cb, cn, idx);
  vq_gather  <<<N_ROWS / 64, 256, 0, stream>>>(z, cb, idx, out, partial);
  vq_finalize<<<1, 256, 0, stream>>>(partial, out + (size_t)N_ROWS * D_DIM);
}
